// MambaModel_15023795601433
// MI455X (gfx1250) — matrix operand order repaired and checked by reading, not compiled, not run on hardware
//
#include <hip/hip_runtime.h>
#include <hip/hip_bf16.h>
#include <cstddef>

// ---------------------------------------------------------------------------
// Mamba forward for MI455X (gfx1250, wave32, WMMA).
//   B=16, L=512, d_model=128, d_inner=256, d_state=32, dt_rank=8, 6 blocks.
// GEMMs: one block = one 16-row m-tile x 8 n-tiles. The A tile (16 x K) is
// cooperatively staged in LDS as f16 (8x reuse across waves, half the bytes),
// waves feed v_wmma_f32_16x16x32_f16 from ds_load_b128 (A) + global b128 (B).
// Scan: one wave per (batch, channel), lane = state index (32 = wave32),
// per-step cross-lane sum via shfl_xor butterfly.
// ---------------------------------------------------------------------------

#define BSZ     16
#define LSEQ    512
#define DMODEL  128
#define DINNER  256
#define DSTATE  32
#define DTRANK  8
#define NBLOCKS 6
#define CONVK   7
#define DCONV   8
#define MROWS   (BSZ * LSEQ)          // 8192
#define XDBL_N  (DTRANK + 2 * DSTATE) // 72
#define XDBL_NP 80                    // padded to multiple of 16

typedef __attribute__((ext_vector_type(16))) _Float16 v16h;
typedef __attribute__((ext_vector_type(8)))  _Float16 v8h;
typedef __attribute__((ext_vector_type(8)))  float    v8f;

// ------------------------------- helpers -----------------------------------

__device__ __forceinline__ float silu_f(float x) {
    return x / (1.0f + __expf(-x));
}
__device__ __forceinline__ float softplus_f(float x) {
    return (x > 20.0f) ? x : log1pf(__expf(x));
}
__device__ __forceinline__ void cvt8(v16h& d, int base, float4 a, float4 b) {
    d[base + 0] = (_Float16)a.x; d[base + 1] = (_Float16)a.y;
    d[base + 2] = (_Float16)a.z; d[base + 3] = (_Float16)a.w;
    d[base + 4] = (_Float16)b.x; d[base + 5] = (_Float16)b.y;
    d[base + 6] = (_Float16)b.z; d[base + 7] = (_Float16)b.w;
}

// ------------------------- 1) pre-conv + exact GELU -------------------------
__global__ __launch_bounds__(256) void k_preconv_gelu(
    const float* __restrict__ x, const float* __restrict__ w,
    const float* __restrict__ bias, float* __restrict__ h)
{
    int idx = blockIdx.x * blockDim.x + threadIdx.x;     // B*L*128 threads
    if (idx >= MROWS * DMODEL) return;
    int c  = idx & (DMODEL - 1);
    int bl = idx >> 7;
    int b  = bl >> 9;
    int l  = bl & (LSEQ - 1);
    float s = bias[c];
    #pragma unroll
    for (int k = 0; k < CONVK; ++k) {
        int ll = l + k - (CONVK / 2);
        if (ll >= 0 && ll < LSEQ) s += w[c * CONVK + k] * x[b * LSEQ + ll];
    }
    h[(size_t)bl * DMODEL + c] = 0.5f * s * (1.0f + erff(s * 0.70710678118f));
}

// ------------------------------ 2) LayerNorm --------------------------------
__global__ __launch_bounds__(256) void k_layernorm(
    const float* __restrict__ h, const float* __restrict__ g,
    const float* __restrict__ bb, float* __restrict__ y)
{
    int row = blockIdx.x * blockDim.x + threadIdx.x;
    if (row >= MROWS) return;
    const float* xr = h + (size_t)row * DMODEL;
    float m = 0.f;
    #pragma unroll 8
    for (int c = 0; c < DMODEL; ++c) m += xr[c];
    m *= (1.0f / DMODEL);
    float v = 0.f;
    #pragma unroll 8
    for (int c = 0; c < DMODEL; ++c) { float d = xr[c] - m; v += d * d; }
    v *= (1.0f / DMODEL);
    float inv = rsqrtf(v + 1e-5f);
    float* yr = y + (size_t)row * DMODEL;
    #pragma unroll 8
    for (int c = 0; c < DMODEL; ++c) yr[c] = (xr[c] - m) * inv * g[c] + bb[c];
}

// ------------------------- 3) WMMA GEMM: C = A * W^T ------------------------
// A: (M, KDIM) f32.  W: (Npad, KDIM) f32 row-major (rows >= N zero-padded).
// C: (M, N) f32.  KDIM % 32 == 0 (compile-time).
// Block = one m-tile x up to 8 n-tiles (8 waves).  A tile staged in LDS (f16).
// Fragment layouts used here (wave32, contiguous k for A and B):
//   A 16x32 : lane m=l&15; elems 0-7  -> K = k0 + 16*(l>=16) + i
//                          elems 8-15 -> K = k0 + 16*(l>=16) + i
//   B 32x16 : lane n=l&15; elem i     -> K = k0 + 16*(l>=16) + i
//   C 16x16 : vgpr j, lane l          -> (M = j + 8*(l>=16), N = l&15)
template <int KDIM, bool ADDRES>
__global__ __launch_bounds__(256) void k_gemm_wmma(
    const float* __restrict__ A,
    const float* __restrict__ W,
    const float* __restrict__ addsrc,
    float* __restrict__ C,
    int M, int N, int Npad)
{
    const int nTiles = Npad >> 4;
    const int nGrp   = (nTiles + 7) >> 3;        // blocks per m-tile
    const int mt     = blockIdx.x / nGrp;
    const int grp    = blockIdx.x % nGrp;
    const int nt     = grp * 8 + (threadIdx.x >> 5);
    const int lane   = threadIdx.x & 31;
    const int lr     = lane & 15;
    const int hi     = lane >> 4;                // 0 or 1

    __shared__ _Float16 As[16 * KDIM];           // 16 rows x KDIM, f16

    // Cooperative A-tile stage: 256 threads, float4 granularity, cvt to f16.
    const float* Abase = A + (size_t)(mt * 16) * KDIM;
    #pragma unroll
    for (int i4 = threadIdx.x; i4 < (16 * KDIM) / 4; i4 += 256) {
        float4 v = *(const float4*)(Abase + i4 * 4);
        As[i4 * 4 + 0] = (_Float16)v.x;
        As[i4 * 4 + 1] = (_Float16)v.y;
        As[i4 * 4 + 2] = (_Float16)v.z;
        As[i4 * 4 + 3] = (_Float16)v.w;
    }
    __syncthreads();

    if (nt >= nTiles) return;                    // wave-uniform (after barrier)

    const float*    Wrow = W  + (size_t)(nt * 16 + lr) * KDIM;
    const _Float16* Alds = As + lr * KDIM;

    v8f acc = {};
    #pragma unroll
    for (int k0 = 0; k0 < KDIM; k0 += 32) {
        // A fragment from LDS: two aligned 16-byte runs
        v8h alo = *(const v8h*)(Alds + k0 + hi * 16);
        v8h ahi = *(const v8h*)(Alds + k0 + hi * 16 + 8);
        // B fragment from global: one 16-float run
        const float4 b0 = *(const float4*)(Wrow + k0 + hi * 16);
        const float4 b1 = *(const float4*)(Wrow + k0 + hi * 16 + 4);
        const float4 b2 = *(const float4*)(Wrow + k0 + hi * 16 + 8);
        const float4 b3 = *(const float4*)(Wrow + k0 + hi * 16 + 12);

        v16h af, bf;
        #pragma unroll
        for (int i = 0; i < 8; ++i) { af[i] = alo[i]; af[i + 8] = ahi[i]; }
        cvt8(bf, 0, b0, b1); cvt8(bf, 8, b2, b3);

        acc = __builtin_amdgcn_wmma_f32_16x16x32_f16(
                  false, af, false, bf, (short)0, acc, false, false);
    }

    const int col = nt * 16 + lr;
    if (col < N) {
        #pragma unroll
        for (int j = 0; j < 8; ++j) {
            int row = mt * 16 + j + hi * 8;
            size_t off = (size_t)row * N + col;
            float v = acc[j];
            if (ADDRES) v += addsrc[off];
            C[off] = v;
        }
    }
}

// -------------------- 3b) weight zero-pad (for x_proj) ----------------------
__global__ __launch_bounds__(256) void k_pad_w(
    const float* __restrict__ src, float* __restrict__ dst, int rows, int k)
{
    int idx = blockIdx.x * blockDim.x + threadIdx.x;      // Npad*k threads
    int r = idx / k, c = idx - r * k;
    dst[idx] = (r < rows) ? src[(size_t)r * k + c] : 0.0f;
}

// ------------- 3c) dt_proj (K=8) + bias + softplus, plain VALU --------------
__global__ __launch_bounds__(256) void k_dtproj_softplus(
    const float* __restrict__ xdbl, const float* __restrict__ dtw,
    const float* __restrict__ dtb, float* __restrict__ dlt)
{
    int idx = blockIdx.x * blockDim.x + threadIdx.x;      // M * 256 threads
    if (idx >= MROWS * DINNER) return;
    int n   = idx & (DINNER - 1);
    int row = idx >> 8;
    const float* xr = xdbl + (size_t)row * XDBL_N;        // dt slice: cols 0..7
    const float* wr = dtw + (size_t)n * DTRANK;
    float s = dtb[n];
    #pragma unroll
    for (int k = 0; k < DTRANK; ++k) s += xr[k] * wr[k];
    dlt[(size_t)row * DINNER + n] = softplus_f(s);
}

// ---------------- 4) depthwise causal conv (D_CONV=8) + SiLU ----------------
__global__ __launch_bounds__(256) void k_dwconv_silu(
    const float* __restrict__ xz, const float* __restrict__ cw,
    const float* __restrict__ cb, float* __restrict__ xc)
{
    int idx = blockIdx.x * blockDim.x + threadIdx.x;      // B*L*256 threads
    if (idx >= MROWS * DINNER) return;
    int d  = idx & (DINNER - 1);
    int bl = idx >> 8;
    int b  = bl >> 9;
    int l  = bl & (LSEQ - 1);
    float s = cb[d];
    #pragma unroll
    for (int k = 0; k < DCONV; ++k) {
        int ll = l + k - (DCONV - 1);                     // causal: left pad 7
        if (ll >= 0)
            s += cw[d * DCONV + k] * xz[((size_t)(b * LSEQ + ll)) * (2 * DINNER) + d];
    }
    xc[(size_t)bl * DINNER + d] = silu_f(s);
}

// --------------------- 5) selective scan (fused epilogue) -------------------
// One wave per (b, d); lane = state index n (DSTATE == wave32 lanes).
// Each lane runs an independent scalar recurrence; y_t = sum over lanes via
// shfl_xor butterfly.  delta/u/z loads are wave-uniform broadcasts; B_t/C_t
// loads are 32-lane coalesced runs.
__global__ __launch_bounds__(256) void k_selective_scan(
    const float* __restrict__ delta, const float* __restrict__ xc,
    const float* __restrict__ xdbl,  const float* __restrict__ Alog,
    const float* __restrict__ Dp,    const float* __restrict__ xz,
    float* __restrict__ ygated)
{
    const int wid  = blockIdx.x * (blockDim.x >> 5) + (threadIdx.x >> 5);
    if (wid >= BSZ * DINNER) return;             // wave-uniform
    const int lane = threadIdx.x & 31;           // n = lane
    const int b    = wid >> 8;
    const int d    = wid & (DINNER - 1);

    const float An = -__expf(Alog[d * DSTATE + lane]);
    const float Dd = Dp[d];
    float hn = 0.0f;

    for (int t = 0; t < LSEQ; ++t) {
        const size_t r  = (size_t)(b * LSEQ + t);
        const float dt  = delta[r * DINNER + d];            // broadcast
        const float u   = xc[r * DINNER + d];               // broadcast
        const float Bn  = xdbl[r * XDBL_N + DTRANK + lane]; // coalesced
        const float Cn  = xdbl[r * XDBL_N + DTRANK + DSTATE + lane];

        hn = __expf(dt * An) * hn + (dt * u) * Bn;
        float y = hn * Cn;
        #pragma unroll
        for (int off = 16; off > 0; off >>= 1)
            y += __shfl_xor(y, off, 32);                    // lane-sum

        if (lane == 0) {
            const float z = xz[r * (2 * DINNER) + DINNER + d];
            ygated[r * DINNER + d] = (y + u * Dd) * silu_f(z);
        }
    }
}

// ------------------------------ 6) output head ------------------------------
__global__ __launch_bounds__(256) void k_head(
    const float* __restrict__ h, const float* __restrict__ ow,
    const float* __restrict__ ob, float* __restrict__ out)
{
    int row = blockIdx.x * blockDim.x + threadIdx.x;
    if (row >= MROWS) return;
    float s = ob[0];
    #pragma unroll 8
    for (int c = 0; c < DMODEL; ++c) s += h[(size_t)row * DMODEL + c] * ow[c];
    out[row] = s;
}

// --------------------------------- driver -----------------------------------

extern "C" void kernel_launch(void* const* d_in, const int* in_sizes, int n_in,
                              void* d_out, int out_size, void* d_ws, size_t ws_size,
                              hipStream_t stream) {
    (void)in_sizes; (void)n_in; (void)out_size; (void)ws_size;

    const float* x    = (const float*)d_in[0];   // (B, L, 1)
    const float* pcw  = (const float*)d_in[1];   // (128, 1, 7)
    const float* pcb  = (const float*)d_in[2];   // (128)
    const float* lng  = (const float*)d_in[3];   // (6, 128)
    const float* lnb  = (const float*)d_in[4];   // (6, 128)
    const float* inw  = (const float*)d_in[5];   // (6, 512, 128)
    const float* cw   = (const float*)d_in[6];   // (6, 256, 8)
    const float* cb   = (const float*)d_in[7];   // (6, 256)
    const float* xpw  = (const float*)d_in[8];   // (6, 72, 256)
    const float* dtw  = (const float*)d_in[9];   // (6, 256, 8)
    const float* dtb  = (const float*)d_in[10];  // (6, 256)
    const float* alog = (const float*)d_in[11];  // (6, 256, 32)
    const float* Dv   = (const float*)d_in[12];  // (6, 256)
    const float* opw  = (const float*)d_in[13];  // (6, 128, 256)
    const float* ow   = (const float*)d_in[14];  // (1, 128)
    const float* ob   = (const float*)d_in[15];  // (1)
    float* out = (float*)d_out;                  // (B, L)

    // workspace layout (f32)
    float* ws     = (float*)d_ws;
    float* hbuf   = ws;                                  // 8192*128
    float* ybuf   = hbuf  + (size_t)MROWS * DMODEL;      // 8192*128
    float* xz     = ybuf  + (size_t)MROWS * DMODEL;      // 8192*512
    float* xc     = xz    + (size_t)MROWS * 2 * DINNER;  // 8192*256
    float* xdbl   = xc    + (size_t)MROWS * DINNER;      // 8192*72
    float* dlt    = xdbl  + (size_t)MROWS * XDBL_N;      // 8192*256
    float* ygated = dlt   + (size_t)MROWS * DINNER;      // 8192*256
    float* wpad   = ygated + (size_t)MROWS * DINNER;     // 80*256

    auto gemmGrid = [](int M, int Npad) {
        int nTiles = Npad >> 4;
        return (M >> 4) * ((nTiles + 7) >> 3);           // m-tiles x n-groups
    };

    // pre-conv + GELU
    k_preconv_gelu<<<(MROWS * DMODEL) / 256, 256, 0, stream>>>(x, pcw, pcb, hbuf);

    for (int i = 0; i < NBLOCKS; ++i) {
        // LayerNorm
        k_layernorm<<<MROWS / 256, 256, 0, stream>>>(
            hbuf, lng + i * DMODEL, lnb + i * DMODEL, ybuf);

        // in_proj: (8192,128) x (512,128)^T -> xz (8192,512)
        k_gemm_wmma<DMODEL, false><<<gemmGrid(MROWS, 2 * DINNER), 256, 0, stream>>>(
            ybuf, inw + (size_t)i * 2 * DINNER * DMODEL,
            nullptr, xz, MROWS, 2 * DINNER, 2 * DINNER);

        // depthwise causal conv + SiLU -> xc (8192,256)
        k_dwconv_silu<<<(MROWS * DINNER) / 256, 256, 0, stream>>>(
            xz, cw + (size_t)i * DINNER * DCONV, cb + (size_t)i * DINNER, xc);

        // pad x_proj weights (72,256) -> (80,256), rows >=72 zero
        k_pad_w<<<(XDBL_NP * DINNER) / 256, 256, 0, stream>>>(
            xpw + (size_t)i * XDBL_N * DINNER, wpad, XDBL_N, DINNER);

        // x_proj: (8192,256) x (80,256)^T -> xdbl (8192,72)
        k_gemm_wmma<DINNER, false><<<gemmGrid(MROWS, XDBL_NP), 256, 0, stream>>>(
            xc, wpad, nullptr, xdbl, MROWS, XDBL_N, XDBL_NP);

        // dt_proj (K=8) + bias + softplus -> delta (8192,256)
        k_dtproj_softplus<<<(MROWS * DINNER) / 256, 256, 0, stream>>>(
            xdbl, dtw + (size_t)i * DINNER * DTRANK, dtb + (size_t)i * DINNER, dlt);

        // selective scan + (+u*D) + gate by silu(z): one wave per (b,d)
        k_selective_scan<<<(BSZ * DINNER * 32) / 256, 256, 0, stream>>>(
            dlt, xc, xdbl, alog + (size_t)i * DINNER * DSTATE,
            Dv + (size_t)i * DINNER, xz, ygated);

        // out_proj + residual (in place on hbuf)
        k_gemm_wmma<DINNER, true><<<gemmGrid(MROWS, DMODEL), 256, 0, stream>>>(
            ygated, opw + (size_t)i * DMODEL * DINNER,
            hbuf, hbuf, MROWS, DMODEL, DMODEL);
    }

    // output head
    k_head<<<MROWS / 256, 256, 0, stream>>>(hbuf, ow, ob, out);
}
